// MSA_47442208752100
// MI455X (gfx1250) — matrix operand order repaired and checked by reading, not compiled, not run on hardware
//
#include <hip/hip_runtime.h>
#include <hip/hip_bf16.h>

typedef __attribute__((ext_vector_type(16))) _Float16 v16h;
typedef __attribute__((ext_vector_type(8)))  _Float16 v8h;
typedef __attribute__((ext_vector_type(8)))  float    v8f;
typedef unsigned int  u32x4 __attribute__((ext_vector_type(4)));
typedef int           i32x4 __attribute__((ext_vector_type(4)));
typedef int           i32x8 __attribute__((ext_vector_type(8)));

#define HEADS 16
#define DIM   64
#define B_    2
#define T_    2048
#define E_    1024
#define F3    3072          /* 3*DIM*HEADS */
#define ATT_SCALE 0.125f    /* DIM^-0.5 */
#define NCH   (T_ / 32)     /* key chunks in attention */

// ---------------- WMMA fragment helpers (layouts per CDNA5 ISA 7.12.2) ----

// A-matrix 16x32 f16: lane L<16 holds row M=L, K = {kb..kb+7, kb+16..kb+23};
// lane L>=16 holds row M=L-16, K = {kb+8..kb+15, kb+24..kb+31}.
static __device__ __forceinline__ v16h load_a_frag(const _Float16* __restrict__ base,
                                                   int ld, int kb) {
    const int lane = threadIdx.x & 31;
    const int row  = lane & 15;
    const int hi   = lane >> 4;
    const _Float16* p = base + (size_t)row * ld + kb + hi * 8;
    v8h lo = *(const v8h*)(p);
    v8h hh = *(const v8h*)(p + 16);
    v16h r;
#pragma unroll
    for (int i = 0; i < 8; ++i) { r[i] = lo[i]; r[i + 8] = hh[i]; }
    return r;
}

// B-matrix 32x16 f16 (K x N): lane L holds column N=L%16,
// K = kb + {8*hi..8*hi+7, 16+8*hi..16+8*hi+7} (hi = L/16), same order as A.
static __device__ __forceinline__ v16h load_b_frag(const _Float16* __restrict__ base,
                                                   int ld, int kb) {
    const int lane = threadIdx.x & 31;
    const int col  = lane & 15;
    const int hi   = lane >> 4;
    const _Float16* p = base + (size_t)col * ld + kb + hi * 8;
    v8h lo = *(const v8h*)(p);
    v8h hh = *(const v8h*)(p + 16);
    v16h r;
#pragma unroll
    for (int i = 0; i < 8; ++i) { r[i] = lo[i]; r[i + 8] = hh[i]; }
    return r;
}

static __device__ __forceinline__ v8f wmma16(v16h a, v16h b, v8f c) {
    return __builtin_amdgcn_wmma_f32_16x16x32_f16(false, a, false, b, (short)0, c,
                                                  false, false);
}

// ---------------- TDM: 2-D tile DMA global -> LDS (CDNA5 ISA) --
static __device__ __forceinline__ void tdm_load_2d(const void* gptr, unsigned lds_addr,
                                                   unsigned tile_w, unsigned tile_h,
                                                   unsigned row_stride_elems) {
    const unsigned long long ga = (unsigned long long)(uintptr_t)gptr;
    const unsigned td0 = row_stride_elems;
    const unsigned td1 = tile_h;
    u32x4 g0;
    g0.x = 1u;                                                 // count=1 (valid)
    g0.y = lds_addr;                                           // lds_addr (bytes)
    g0.z = (unsigned)ga;                                       // global_addr[31:0]
    g0.w = (unsigned)((ga >> 32) & 0x01FFFFFFu) | (2u << 30);  // addr[56:32] | type=2
    i32x8 g1;
    g1[0] = (int)(1u << 16);                                   // wg_mask=0, data_size=2B
    g1[1] = (int)((td0 & 0xFFFFu) << 16);                      // tensor_dim0[15:0]
    g1[2] = (int)((td0 >> 16) | ((td1 & 0xFFFFu) << 16));      // dim0.hi | dim1.lo
    g1[3] = (int)((td1 >> 16) | ((tile_w & 0xFFFFu) << 16));   // dim1.hi | tile_dim0
    g1[4] = (int)(tile_h & 0xFFFFu);                           // tile_dim1, tile_dim2=0
    g1[5] = (int)row_stride_elems;                             // dim0_stride[31:0]
    g1[6] = 0;
    g1[7] = 0;
    i32x4 z4 = {0, 0, 0, 0};
#if __has_include(<hip/amd_detail/amd_gfx1250_TDM.h>)
    i32x8 z8 = {0, 0, 0, 0, 0, 0, 0, 0};
    __builtin_amdgcn_tensor_load_to_lds(g0, g1, z4, z4, z8, 0);
#else
    __builtin_amdgcn_tensor_load_to_lds(g0, g1, z4, z4, 0);
#endif
}

static __device__ __forceinline__ unsigned lds_off(const void* p) {
    // low 32 bits of a generic pointer into LDS are the LDS byte offset
    return (unsigned)__builtin_amdgcn_readfirstlane((int)(unsigned)(uintptr_t)p);
}

// ---------------- Kernel 0: fp32 -> f16 conversion, 8 elems / thread ---------------

__global__ void cvt_f32_f16(const float* __restrict__ src, _Float16* __restrict__ dst,
                            int n8) {
    const int i = blockIdx.x * blockDim.x + threadIdx.x;
    if (i < n8) {
        const float4 f0 = ((const float4*)src)[i * 2];
        const float4 f1 = ((const float4*)src)[i * 2 + 1];
        v8h h;
        h[0] = (_Float16)f0.x; h[1] = (_Float16)f0.y;
        h[2] = (_Float16)f0.z; h[3] = (_Float16)f0.w;
        h[4] = (_Float16)f1.x; h[5] = (_Float16)f1.y;
        h[6] = (_Float16)f1.z; h[7] = (_Float16)f1.w;
        ((v8h*)dst)[i] = h;
    }
}

// ---------------- shared GEMM core: 32(M) x 64(N) wave tile, ping-pong over k ------

static __device__ __forceinline__ void load_ab(const _Float16* __restrict__ Ab0,
                                               const _Float16* __restrict__ Ab1,
                                               const _Float16* const Bb[4], int k,
                                               v16h& a0, v16h& a1, v16h b[4]) {
    a0 = load_a_frag(Ab0, E_, k);
    a1 = load_a_frag(Ab1, E_, k);
#pragma unroll
    for (int nt = 0; nt < 4; ++nt) b[nt] = load_b_frag(Bb[nt], E_, k);
}

static __device__ __forceinline__ void gemm_step(v8f acc[8], const v16h& a0,
                                                 const v16h& a1, const v16h b[4]) {
#pragma unroll
    for (int nt = 0; nt < 4; ++nt) {
        acc[nt]     = wmma16(a0, b[nt], acc[nt]);
        acc[4 + nt] = wmma16(a1, b[nt], acc[4 + nt]);
    }
}

static __device__ __forceinline__ void gemm_mainloop(const _Float16* __restrict__ Ab0,
                                                     const _Float16* __restrict__ Ab1,
                                                     const _Float16* const Bb[4],
                                                     v8f acc[8]) {
    v16h aA0, aA1, bA[4], aB0, aB1, bB[4];
    load_ab(Ab0, Ab1, Bb, 0, aA0, aA1, bA);
    for (int k = 0; k < E_ - 64; k += 64) {          // ping-pong: no register copies
        load_ab(Ab0, Ab1, Bb, k + 32, aB0, aB1, bB);
        gemm_step(acc, aA0, aA1, bA);
        load_ab(Ab0, Ab1, Bb, k + 64, aA0, aA1, bA);
        gemm_step(acc, aB0, aB1, bB);
    }
    load_ab(Ab0, Ab1, Bb, E_ - 32, aB0, aB1, bB);
    gemm_step(acc, aA0, aA1, bA);
    gemm_step(acc, aB0, aB1, bB);
}

// ---------------- Kernel 1: QKV projection GEMM + layout scatter -------------------
// out[m, f] = sum_k x[m,k] * w_qkv[f,k];  f = kk*1024 + d*16 + h  (h fastest).

__global__ __launch_bounds__(128) void qkv_gemm(const _Float16* __restrict__ xh,
                                                const _Float16* __restrict__ wh,
                                                _Float16* __restrict__ Qo,
                                                _Float16* __restrict__ Ko,
                                                _Float16* __restrict__ Vt) {
    const int wid   = blockIdx.x * 4 + (threadIdx.x >> 5);
    const int mtile = wid / 48;          // 0..127 (32 rows each, M = 4096)
    const int ngrp  = wid % 48;          // 4 n-tiles each -> N = 3072
    const int lane  = threadIdx.x & 31;
    const int col = lane & 15, hi = lane >> 4;

    const _Float16* Ab0 = xh + (size_t)mtile * 32 * E_;
    const _Float16* Ab1 = Ab0 + 16 * E_;
    const _Float16* Bb[4];
#pragma unroll
    for (int nt = 0; nt < 4; ++nt) Bb[nt] = wh + (size_t)(ngrp * 4 + nt) * 16 * E_;

    v8f acc[8] = {};
    gemm_mainloop(Ab0, Ab1, Bb, acc);

    // scatter: 32-row tile never crosses the batch boundary (2048 % 32 == 0)
    const int m0 = mtile * 32;
    const int bI = m0 >> 11;
    const int bh = bI * HEADS + col;     // col == head index
#pragma unroll
    for (int mt = 0; mt < 2; ++mt) {
        const int t0 = (m0 & 2047) + mt * 16 + hi * 8;
#pragma unroll
        for (int nt = 0; nt < 4; ++nt) {
            const int ntile = ngrp * 4 + nt;
            const int kk = ntile >> 6;       // q / k / v
            const int dd = ntile & 63;       // head-dim index
            _Float16* pq = Qo + ((size_t)bh * T_ + t0) * DIM + dd;
            _Float16* pk = Ko + ((size_t)bh * T_ + t0) * DIM + dd;
            _Float16* pv = Vt + ((size_t)bh * DIM + dd) * T_ + t0;
#pragma unroll
            for (int r = 0; r < 8; ++r) {
                const _Float16 v = (_Float16)acc[mt * 4 + nt][r];
                if (kk == 0)      pq[(size_t)r * DIM] = v;
                else if (kk == 1) pk[(size_t)r * DIM] = v;
                else              pv[r] = v;
            }
        }
    }
}

// ---------------- Kernel 2: flash attention, TDM-staged K/V -----------------------
// 2 waves / block; each wave owns a 16-row Q tile and double-buffered
// K (32x64) / V^T (64x32) LDS tiles filled by tensor_load_to_lds.

__global__ __launch_bounds__(64) void attn_kernel(const _Float16* __restrict__ Q,
                                                  const _Float16* __restrict__ K,
                                                  const _Float16* __restrict__ Vt,
                                                  _Float16* __restrict__ O) {
    __shared__ __align__(32) _Float16 Kl[2][2][32 * 64];   // [wave][buf] 16 KB
    __shared__ __align__(32) _Float16 Vl[2][2][64 * 32];   // [wave][buf] 16 KB
    __shared__ __align__(32) _Float16 Pl[2][16 * 32];      // P staging     2 KB

    const int wslot = threadIdx.x >> 5;
    const int bh    = blockIdx.x >> 6;          // 0..31
    const int qg    = blockIdx.x & 63;
    const int qt    = qg * 2 + wslot;           // Q row-tile within (b,h)
    const int lane  = threadIdx.x & 31;
    const int col = lane & 15, hi = lane >> 4;

    const _Float16* Kb = K  + (size_t)bh * T_ * DIM;
    const _Float16* Vb = Vt + (size_t)bh * DIM * T_;

    const unsigned klA = lds_off(&Kl[wslot][0][0]), klB = lds_off(&Kl[wslot][1][0]);
    const unsigned vlA = lds_off(&Vl[wslot][0][0]), vlB = lds_off(&Vl[wslot][1][0]);
    const _Float16* KcA = &Kl[wslot][0][0];
    const _Float16* KcB = &Kl[wslot][1][0];
    const _Float16* VcA = &Vl[wslot][0][0];
    const _Float16* VcB = &Vl[wslot][1][0];

    // stage chunks 0 and 1 (each: K 32x64 row-major, V^T 64 rows x 32 cols)
    tdm_load_2d(Kb,            klA, 64, 32, 64);
    tdm_load_2d(Vb,            vlA, 32, 64, T_);
    tdm_load_2d(Kb + 32 * DIM, klB, 64, 32, 64);
    tdm_load_2d(Vb + 32,       vlB, 32, 64, T_);

    // Q fragments, pre-scaled by 1/sqrt(D)
    const _Float16* Qbase = Q + ((size_t)bh * T_ + qt * 16) * DIM;
    v16h aq0 = load_a_frag(Qbase, DIM, 0);
    v16h aq1 = load_a_frag(Qbase, DIM, 32);
    const _Float16 sc = (_Float16)ATT_SCALE;
#pragma unroll
    for (int i = 0; i < 16; ++i) { aq0[i] *= sc; aq1[i] *= sc; }

    v8f o0 = {}, o1 = {}, o2 = {}, o3 = {};
    float mrow[8], lrow[8];
#pragma unroll
    for (int r = 0; r < 8; ++r) { mrow[r] = -3.0e38f; lrow[r] = 0.0f; }

    _Float16* pb = Pl[wslot];

    auto process = [&](int c, const _Float16* Kc, const _Float16* Vc,
                       unsigned klX, unsigned vlX) {
        if (c < NCH - 1) __builtin_amdgcn_s_wait_tensorcnt(2);  // chunk c landed
        else             __builtin_amdgcn_s_wait_tensorcnt(0);

        // ---- S = Q K^T for 32 keys (two 16-wide n-tiles, K-dim 64) --------------
        v8f s0 = {}, s1 = {};
        s0 = wmma16(aq0, load_b_frag(Kc,            DIM, 0),  s0);
        s0 = wmma16(aq1, load_b_frag(Kc,            DIM, 32), s0);
        s1 = wmma16(aq0, load_b_frag(Kc + 16 * DIM, DIM, 0),  s1);
        s1 = wmma16(aq1, load_b_frag(Kc + 16 * DIM, DIM, 32), s1);

        // ---- online softmax (rows live across 16-lane halves) -------------------
        float corr[8];
#pragma unroll
        for (int r = 0; r < 8; ++r) {
            float tm = fmaxf(s0[r], s1[r]);
            tm = fmaxf(tm, __shfl_xor(tm, 1, 32));
            tm = fmaxf(tm, __shfl_xor(tm, 2, 32));
            tm = fmaxf(tm, __shfl_xor(tm, 4, 32));
            tm = fmaxf(tm, __shfl_xor(tm, 8, 32));
            const float mn = fmaxf(mrow[r], tm);
            corr[r] = __expf(mrow[r] - mn);
            mrow[r] = mn;
            const float p0 = __expf(s0[r] - mn);
            const float p1 = __expf(s1[r] - mn);
            s0[r] = p0; s1[r] = p1;
            float rs = p0 + p1;
            rs += __shfl_xor(rs, 1, 32);
            rs += __shfl_xor(rs, 2, 32);
            rs += __shfl_xor(rs, 4, 32);
            rs += __shfl_xor(rs, 8, 32);
            lrow[r] = lrow[r] * corr[r] + rs;
        }
#pragma unroll
        for (int r = 0; r < 8; ++r) {
            o0[r] *= corr[r]; o1[r] *= corr[r];
            o2[r] *= corr[r]; o3[r] *= corr[r];
        }

        // ---- P (C-layout) -> LDS 16x32 f16 -> A-fragment ------------------------
#pragma unroll
        for (int r = 0; r < 8; ++r) {
            const int row = r + hi * 8;
            pb[row * 32 + col]      = (_Float16)s0[r];
            pb[row * 32 + col + 16] = (_Float16)s1[r];
        }
        __asm__ volatile("s_wait_dscnt 0" ::: "memory");   // wave-local DS ordering
        v16h ap = load_a_frag(pb, 32, 0);

        // ---- O += P V  (V^T rows are contiguous B-fragments) --------------------
        o0 = wmma16(ap, load_b_frag(Vc + 0 * 16 * 32, 32, 0), o0);
        o1 = wmma16(ap, load_b_frag(Vc + 1 * 16 * 32, 32, 0), o1);
        o2 = wmma16(ap, load_b_frag(Vc + 2 * 16 * 32, 32, 0), o2);
        o3 = wmma16(ap, load_b_frag(Vc + 3 * 16 * 32, 32, 0), o3);

        // ---- refill this buffer with chunk c+2 once all DS reads retired --------
        if (c + 2 < NCH) {
            __asm__ volatile("s_wait_dscnt 0" ::: "memory");
            const int j2 = (c + 2) * 32;
            tdm_load_2d(Kb + (size_t)j2 * DIM, klX, 64, 32, 64);
            tdm_load_2d(Vb + j2,               vlX, 32, 64, T_);
        }
    };

    for (int c2 = 0; c2 < NCH / 2; ++c2) {      // x2 unroll: static buffer selection
        process(c2 * 2,     KcA, VcA, klA, vlA);
        process(c2 * 2 + 1, KcB, VcB, klB, vlB);
    }

    // ---- normalize, emit [B, T, H*D] f16 for output projection ------------------
    const int bI = bh >> 4, h = bh & 15;
#pragma unroll
    for (int r = 0; r < 8; ++r) {
        const float inv = 1.0f / lrow[r];
        const int t = qt * 16 + r + hi * 8;
        _Float16* dst = O + ((size_t)bI * T_ + t) * (HEADS * DIM) + h * DIM + col;
        dst[0]  = (_Float16)(o0[r] * inv);
        dst[16] = (_Float16)(o1[r] * inv);
        dst[32] = (_Float16)(o2[r] * inv);
        dst[48] = (_Float16)(o3[r] * inv);
    }
}

// ---------------- Kernel 3: output projection GEMM, fp32 result --------------------
// out[m, e] = sum_f attn[m, f] * w_out[e, f].  Wave tile 32x64, ping-pong k loop.

__global__ __launch_bounds__(128) void out_gemm(const _Float16* __restrict__ Ah,
                                                const _Float16* __restrict__ Wh,
                                                float* __restrict__ out) {
    const int wid   = blockIdx.x * 4 + (threadIdx.x >> 5);
    const int mtile = wid >> 4;          // 0..127 (32 rows each)
    const int ngrp  = wid & 15;          // 4 n-tiles each -> N = 1024
    const int lane  = threadIdx.x & 31;
    const int col = lane & 15, hi = lane >> 4;

    const _Float16* Ab0 = Ah + (size_t)mtile * 32 * E_;
    const _Float16* Ab1 = Ab0 + 16 * E_;
    const _Float16* Bb[4];
#pragma unroll
    for (int nt = 0; nt < 4; ++nt) Bb[nt] = Wh + (size_t)(ngrp * 4 + nt) * 16 * E_;

    v8f acc[8] = {};
    gemm_mainloop(Ab0, Ab1, Bb, acc);

#pragma unroll
    for (int mt = 0; mt < 2; ++mt) {
#pragma unroll
        for (int nt = 0; nt < 4; ++nt) {
            const int n = (ngrp * 4 + nt) * 16 + col;
            float* pd = out + (size_t)(mtile * 32 + mt * 16 + hi * 8) * E_ + n;
#pragma unroll
            for (int r = 0; r < 8; ++r) pd[(size_t)r * E_] = acc[mt * 4 + nt][r];
        }
    }
}

// ---------------- host side --------------------------------------------------------

extern "C" void kernel_launch(void* const* d_in, const int* in_sizes, int n_in,
                              void* d_out, int out_size, void* d_ws, size_t ws_size,
                              hipStream_t stream) {
    const float* x    = (const float*)d_in[0];
    const float* wqkv = (const float*)d_in[1];
    const float* wout = (const float*)d_in[2];
    float* out = (float*)d_out;

    char* ws = (char*)d_ws;
    const size_t MB = 1024 * 1024;
    _Float16* xh    = (_Float16*)(ws + 0);        //  8 MB  x as f16 [4096,1024]
    _Float16* wqkvh = (_Float16*)(ws + 8  * MB);  //  6 MB  w_qkv f16 [3072,1024]
    _Float16* wouth = (_Float16*)(ws + 14 * MB);  //  2 MB  w_out f16 [1024,1024]
    _Float16* Qb    = (_Float16*)(ws + 16 * MB);  //  8 MB  Q  [BH,T,D]
    _Float16* Kb    = (_Float16*)(ws + 24 * MB);  //  8 MB  K  [BH,T,D]
    _Float16* Vtb   = (_Float16*)(ws + 32 * MB);  //  8 MB  V^T [BH,D,T]
    _Float16* Ao    = (_Float16*)(ws + 40 * MB);  //  8 MB  attn out [B,T,H*D]

    cvt_f32_f16<<<(B_ * T_ * E_ / 8 + 255) / 256, 256, 0, stream>>>(x, xh, B_ * T_ * E_ / 8);
    cvt_f32_f16<<<(F3 * E_ / 8 + 255) / 256, 256, 0, stream>>>(wqkv, wqkvh, F3 * E_ / 8);
    cvt_f32_f16<<<(E_ * E_ / 8 + 255) / 256, 256, 0, stream>>>(wout, wouth, E_ * E_ / 8);

    // 128 m-tiles * 48 n-groups = 6144 waves / 4 per block
    qkv_gemm<<<1536, 128, 0, stream>>>(xh, wqkvh, Qb, Kb, Vtb);
    // 32 (b,h) * 64 q-groups, 2 waves per block
    attn_kernel<<<2048, 64, 0, stream>>>(Qb, Kb, Vtb, Ao);
    // 128 m-tiles * 16 n-groups = 2048 waves / 4 per block
    out_gemm<<<512, 128, 0, stream>>>(Ao, wouth, out);
}
